// StructureModule_71519795413185
// MI455X (gfx1250) — compile-verified, matrix operand order checked by reading, not run on hardware
//
#include <hip/hip_runtime.h>
#include <math.h>

// ---- problem constants (match reference) ----
#define NL 8     // layers
#define DM 512   // model dim
#define PP 128   // pair dim
#define NH 8     // heads
#define HD 64    // head dim
#define NB 2     // batch
#define NS 4     // msa seqs
#define SS 384   // sequence length

typedef __attribute__((ext_vector_type(16))) __bf16 v16bf;
typedef __attribute__((ext_vector_type(8)))  float  v8f;
typedef __attribute__((ext_vector_type(4)))  float  v4f;
typedef __attribute__((ext_vector_type(4)))  unsigned int v4u;
typedef __attribute__((ext_vector_type(8)))  unsigned int v8u;

// ---------------- WMMA fragment helpers (CDNA5 16x16x32 bf16 layout) -------
// Per lane: row/col = lane&15, khalf = lane>>4.
// Fragment = elements [khalf*8 .. khalf*8+7] and [16+khalf*8 .. 16+khalf*8+7]
// of that row -> two contiguous 16-byte loads.
static __device__ __forceinline__ v16bf frag_ld(const __bf16* p, int khalf) {
    const v4u* p0 = (const v4u*)(p + khalf * 8);
    const v4u* p1 = (const v4u*)(p + 16 + khalf * 8);
    v4u a = *p0, b = *p1;
    v8u u;
    u[0] = a[0]; u[1] = a[1]; u[2] = a[2]; u[3] = a[3];
    u[4] = b[0]; u[5] = b[1]; u[6] = b[2]; u[7] = b[3];
    return __builtin_bit_cast(v16bf, u);
}

// f32 source (pair_repr): vector loads + in-register cvt to bf16
static __device__ __forceinline__ v16bf frag_ld_f32(const float* p, int khalf) {
    v4f f0 = *(const v4f*)(p + khalf * 8);
    v4f f1 = *(const v4f*)(p + khalf * 8 + 4);
    v4f f2 = *(const v4f*)(p + 16 + khalf * 8);
    v4f f3 = *(const v4f*)(p + 16 + khalf * 8 + 4);
    v16bf r;
#pragma unroll
    for (int i = 0; i < 4; ++i) {
        r[i]      = (__bf16)f0[i];
        r[4 + i]  = (__bf16)f1[i];
        r[8 + i]  = (__bf16)f2[i];
        r[12 + i] = (__bf16)f3[i];
    }
    return r;
}

static __device__ __forceinline__ v8f wmma_bf16(v16bf a, v16bf b, v8f c) {
    return __builtin_amdgcn_wmma_f32_16x16x32_bf16(false, a, false, b, (short)0, c,
                                                   false, false);
}

// ---------------- LDS-staged block GEMM: C = act(A_bf16 @ W_f32 + bias) ----
// M mult of 128, N mult of 64, K mult of 32. Block 256 thr = 8 wave32 waves,
// each wave computes a 32x32 tile (2x2 WMMA accumulators). B is staged
// N-major (transposed) in LDS so both A and B fragments are b128 loads.
template <int ADDC, int RELU, int WRITE_F32>
__global__ void gemm_bf16_kernel(const __bf16* __restrict__ A, const float* __restrict__ W,
                                 const float* __restrict__ bias,
                                 float* __restrict__ Cf32, __bf16* __restrict__ Cbf16,
                                 int M, int N, int K) {
    __shared__ __bf16 As[128][40];    // [m][k], 80B row stride (16B aligned)
    __shared__ __bf16 Bt[64][40];     // [n][k] transposed
    const int t = threadIdx.x;
    const int lane = t & 31, wid = t >> 5;
    const int wr = wid & 3, wc = wid >> 2;
    const int m0 = blockIdx.x * 128, n0 = blockIdx.y * 64;
    const int khalf = lane >> 4, ln = lane & 15;

    v8f zero = {};
    v8f acc[2][2];
#pragma unroll
    for (int i = 0; i < 2; ++i)
#pragma unroll
        for (int j = 0; j < 2; ++j) acc[i][j] = zero;

    for (int k0 = 0; k0 < K; k0 += 32) {
        // stage A tile 128x32 bf16 as 16B chunks (512 chunks, 2 per thread)
        for (int u = t; u < 128 * 4; u += 256) {
            int r = u >> 2, c8 = (u & 3) * 8;
            *(v4u*)&As[r][c8] = *(const v4u*)(A + (size_t)(m0 + r) * K + k0 + c8);
        }
        // stage W tile 32x64 f32 -> bf16, transposed to [n][k]
        for (int u = t; u < 32 * 64; u += 256) {
            int r = u >> 6, c = u & 63;  // k=r, n=c (coalesced global read)
            Bt[c][r] = (__bf16)W[(size_t)(k0 + r) * N + n0 + c];
        }
        if (k0 + 32 < K) {  // prefetch next tiles into cache
            __builtin_prefetch(W + (size_t)(k0 + 32) * N + n0 + (t & 63), 0, 1);
            __builtin_prefetch(A + (size_t)(m0 + (t >> 1)) * K + k0 + 32, 0, 1);
        }
        __syncthreads();
#pragma unroll
        for (int sm = 0; sm < 2; ++sm) {
            v16bf a = frag_ld(&As[wr * 32 + sm * 16 + ln][0], khalf);
#pragma unroll
            for (int sn = 0; sn < 2; ++sn) {
                v16bf b = frag_ld(&Bt[wc * 32 + sn * 16 + ln][0], khalf);
                acc[sm][sn] = wmma_bf16(a, b, acc[sm][sn]);
            }
        }
        __syncthreads();
    }

#pragma unroll
    for (int sm = 0; sm < 2; ++sm)
#pragma unroll
        for (int sn = 0; sn < 2; ++sn) {
            int col = n0 + wc * 32 + sn * 16 + ln;
            float bv = bias[col];
#pragma unroll
            for (int r = 0; r < 8; ++r) {
                int row = m0 + wr * 32 + sm * 16 + r + 8 * khalf;
                size_t idx = (size_t)row * N + col;
                float v = acc[sm][sn][r] + bv;
                if (ADDC) v += Cf32[idx];
                if (RELU) v = v > 0.f ? v : 0.f;
                if (WRITE_F32) Cf32[idx] = v;
                Cbf16[idx] = (__bf16)v;
            }
        }
}

// -------- pair bias: bias[b,i,j,h] = sum_p pair[b,i,j,p] * Wpb[l,p,h] ------
// M = B*S*S flattened (K=128 contiguous), N = 8 heads padded to one 16-tile.
// Wpb staged N-major in LDS (16 x 128, padded stride 136 -> 272B, 16B-aligned)
__global__ void pair_bias_kernel(const float* __restrict__ pair,
                                 const float* __restrict__ Wpb, int layer,
                                 __bf16* __restrict__ biasOut) {
    __shared__ __bf16 wt[16][136];  // [h][p]
    const int t = threadIdx.x, lane = t & 31, wid = t >> 5;
    for (int u = t; u < 16 * 128; u += 256) {
        int n = u >> 7, p = u & 127;
        wt[n][p] = (__bf16)(n < NH ? Wpb[((size_t)layer * PP + p) * NH + n] : 0.f);
    }
    __syncthreads();
    const int khalf = lane >> 4, ln = lane & 15;
    const size_t m0 = (size_t)blockIdx.x * 128 + (size_t)wid * 16;
    v8f acc = {};
    const float* arow = pair + (m0 + ln) * PP;
#pragma unroll
    for (int ks = 0; ks < PP; ks += 32) {
        v16bf a = frag_ld_f32(arow + ks, khalf);
        v16bf b = frag_ld(&wt[ln][ks], khalf);
        acc = wmma_bf16(a, b, acc);
    }
    if (ln < NH) {
#pragma unroll
        for (int r = 0; r < 8; ++r) {
            size_t row = m0 + r + 8 * khalf;
            biasOut[row * NH + ln] = (__bf16)acc[r];
        }
    }
}

// -------- scores[b,h,i,j] = 0.125 * q_i . k_j + bias[b,i,j,h] + bpb[l,h] ---
// Head slices are contiguous 64 columns of the (B,S,D) buffers, so both A (q)
// and B (k, N-major by construction) fragments are direct global b128 loads.
__global__ void scores_kernel(const __bf16* __restrict__ q, const __bf16* __restrict__ k,
                              const __bf16* __restrict__ biasIn,
                              const float* __restrict__ bpb, int layer,
                              float* __restrict__ scores) {
    const int t = threadIdx.x, lane = t & 31, wid = t >> 5;
    const int khalf = lane >> 4, ln = lane & 15;
    const int bh = blockIdx.z, b = bh >> 3, h = bh & 7;
    const int i0 = blockIdx.x * 16;
    const int j0 = blockIdx.y * 128 + wid * 16;
    v8f acc = {};
    const __bf16* qrow = q + ((size_t)(b * SS) + i0 + ln) * DM + h * HD;
    const __bf16* krow = k + ((size_t)(b * SS) + j0 + ln) * DM + h * HD;
#pragma unroll
    for (int ks = 0; ks < HD; ks += 32) {
        v16bf a  = frag_ld(qrow + ks, khalf);
        v16bf bb = frag_ld(krow + ks, khalf);
        acc = wmma_bf16(a, bb, acc);
    }
    const float pb = bpb[layer * NH + h];
    const int j = j0 + ln;
#pragma unroll
    for (int r = 0; r < 8; ++r) {
        int i = i0 + r + 8 * khalf;
        size_t bidx = (((size_t)b * SS + i) * SS + j) * NH + h;
        float v = acc[r] * 0.125f + (float)biasIn[bidx] + pb;
        scores[(((size_t)(b * NH + h)) * SS + i) * SS + j] = v;
    }
}

// -------- row softmax over 384, one wave per row, probs -> bf16 ------------
// Each lane owns 12 contiguous elements -> 3x b128 loads.
__global__ void softmax_kernel(const float* __restrict__ scores,
                               __bf16* __restrict__ probs) {
    const int t = threadIdx.x, lane = t & 31, wid = t >> 5;
    const size_t row = (size_t)blockIdx.x * 8 + wid;  // < B*H*S
    const float* src = scores + row * SS + lane * 12;
    v4f e0 = *(const v4f*)(src);
    v4f e1 = *(const v4f*)(src + 4);
    v4f e2 = *(const v4f*)(src + 8);
    float mx = -1e30f;
#pragma unroll
    for (int u = 0; u < 4; ++u) mx = fmaxf(mx, fmaxf(e0[u], fmaxf(e1[u], e2[u])));
#pragma unroll
    for (int off = 16; off > 0; off >>= 1) mx = fmaxf(mx, __shfl_xor(mx, off, 32));
    float sum = 0.f;
#pragma unroll
    for (int u = 0; u < 4; ++u) {
        e0[u] = __expf(e0[u] - mx); sum += e0[u];
        e1[u] = __expf(e1[u] - mx); sum += e1[u];
        e2[u] = __expf(e2[u] - mx); sum += e2[u];
    }
#pragma unroll
    for (int off = 16; off > 0; off >>= 1) sum += __shfl_xor(sum, off, 32);
    float inv = 1.f / sum;
    __bf16* dst = probs + row * SS + lane * 12;
#pragma unroll
    for (int u = 0; u < 4; ++u) {
        dst[u]     = (__bf16)(e0[u] * inv);
        dst[4 + u] = (__bf16)(e1[u] * inv);
        dst[8 + u] = (__bf16)(e2[u] * inv);
    }
}

// -------- out[b,i,h,:] = probs[b,h,i,:] @ v[b,:,h,:] -----------------------
// V tile (32 k x 64 d) staged transposed to LDS [d][k]; all 8 waves (over i)
// share it. A fragments load straight from global probs (contiguous rows).
__global__ void attnv_kernel(const __bf16* __restrict__ probs,
                             const __bf16* __restrict__ vv,
                             __bf16* __restrict__ outp) {
    __shared__ __bf16 Vt[64][40];  // [d][k]
    const int t = threadIdx.x, lane = t & 31, wid = t >> 5;
    const int khalf = lane >> 4, ln = lane & 15;
    const int bh = blockIdx.z, b = bh >> 3, h = bh & 7;
    const int i0 = blockIdx.x * 128 + wid * 16;
    v8f zero = {};
    v8f acc[4];
#pragma unroll
    for (int nt = 0; nt < 4; ++nt) acc[nt] = zero;
    const __bf16* prow  = probs + (((size_t)(b * NH + h)) * SS + i0 + ln) * SS;
    const __bf16* vbase = vv + (size_t)(b * SS) * DM + h * HD;
    for (int kk = 0; kk < SS; kk += 32) {
        for (int u = t; u < 32 * 64; u += 256) {
            int r = u >> 6, c = u & 63;  // k=r, d=c (coalesced global read)
            Vt[c][r] = vbase[(size_t)(kk + r) * DM + c];
        }
        __syncthreads();
        v16bf a = frag_ld(prow + kk, khalf);
#pragma unroll
        for (int nt = 0; nt < 4; ++nt) {
            v16bf bb = frag_ld(&Vt[nt * 16 + ln][0], khalf);
            acc[nt] = wmma_bf16(a, bb, acc[nt]);
        }
        __syncthreads();
    }
#pragma unroll
    for (int nt = 0; nt < 4; ++nt) {
        int n = nt * 16 + ln;
#pragma unroll
        for (int r = 0; r < 8; ++r) {
            int i = i0 + r + 8 * khalf;
            outp[((size_t)(b * SS) + i) * DM + h * HD + n] = (__bf16)acc[nt][r];
        }
    }
}

// -------- init x (msa[:,0]) + bf16 mirror + zero coords --------------------
__global__ void init_kernel(const float* __restrict__ msa, float* __restrict__ xf,
                            __bf16* __restrict__ xb, float* __restrict__ coords) {
    size_t i = (size_t)blockIdx.x * blockDim.x + threadIdx.x;
    if (i < (size_t)NB * SS * DM) {
        size_t row = i / DM, d = i % DM;
        size_t b = row / SS, s = row % SS;
        float v = msa[((b * NS) * SS + s) * DM + d];
        xf[i] = v;
        xb[i] = (__bf16)v;
    }
    if (i < (size_t)NB * SS * 3) coords[i] = 0.f;
}

// -------- coords += (h @ Wb2 + bb2)[:, :3] ---------------------------------
__global__ void coord_update_kernel(const __bf16* __restrict__ hbuf,
                                    const float* __restrict__ Wb2,
                                    const float* __restrict__ bb2,
                                    float* __restrict__ coords) {
    int row = blockIdx.x * blockDim.x + threadIdx.x;
    if (row >= NB * SS) return;
    const __bf16* hr = hbuf + (size_t)row * DM;
    float s0 = 0.f, s1 = 0.f, s2 = 0.f;
    for (int k2 = 0; k2 < DM; ++k2) {
        float hv = (float)hr[k2];
        s0 += hv * Wb2[k2 * 6 + 0];
        s1 += hv * Wb2[k2 * 6 + 1];
        s2 += hv * Wb2[k2 * 6 + 2];
    }
    coords[row * 3 + 0] += s0 + bb2[0];
    coords[row * 3 + 1] += s1 + bb2[1];
    coords[row * 3 + 2] += s2 + bb2[2];
}

// -------- confidence = sigmoid(c1 @ Wc2 + bc2) -----------------------------
__global__ void confidence_kernel(const __bf16* __restrict__ c1,
                                  const float* __restrict__ Wc2,
                                  const float* __restrict__ bc2,
                                  float* __restrict__ conf) {
    int row = blockIdx.x * blockDim.x + threadIdx.x;
    if (row >= NB * SS) return;
    const __bf16* cr = c1 + (size_t)row * (DM / 2);
    float z = 0.f;
    for (int k2 = 0; k2 < DM / 2; ++k2) z += (float)cr[k2] * Wc2[k2];
    z += bc2[0];
    conf[row] = 1.f / (1.f + __expf(-z));
}

// ---------------------------------------------------------------------------
extern "C" void kernel_launch(void* const* d_in, const int* in_sizes, int n_in,
                              void* d_out, int out_size, void* d_ws, size_t ws_size,
                              hipStream_t stream) {
    (void)in_sizes; (void)n_in; (void)out_size; (void)ws_size;
    const float* msa  = (const float*)d_in[0];
    const float* pair = (const float*)d_in[1];
    const float* Wq   = (const float*)d_in[2];
    const float* bq   = (const float*)d_in[3];
    const float* Wk   = (const float*)d_in[4];
    const float* bk   = (const float*)d_in[5];
    const float* Wv   = (const float*)d_in[6];
    const float* bv   = (const float*)d_in[7];
    const float* Wpb  = (const float*)d_in[8];
    const float* bpb  = (const float*)d_in[9];
    const float* Wo   = (const float*)d_in[10];
    const float* bo   = (const float*)d_in[11];
    const float* Wb1  = (const float*)d_in[12];
    const float* bb1  = (const float*)d_in[13];
    const float* Wb2  = (const float*)d_in[14];
    const float* bb2  = (const float*)d_in[15];
    const float* Wc1  = (const float*)d_in[16];
    const float* bc1  = (const float*)d_in[17];
    const float* Wc2  = (const float*)d_in[18];
    const float* bc2  = (const float*)d_in[19];
    float* outp = (float*)d_out;

    // ---- workspace carve-out (~26 MB) ----
    char* ws = (char*)d_ws;
    size_t off = 0;
    auto carve = [&](size_t bytes) {
        char* p = ws + off;
        off += (bytes + 255) & ~(size_t)255;
        return p;
    };
    const size_t MR = (size_t)NB * SS;  // 768 rows
    float*  x_f32   = (float*) carve(MR * DM * 4);
    __bf16* x_bf    = (__bf16*)carve(MR * DM * 2);
    __bf16* qbuf    = (__bf16*)carve(MR * DM * 2);
    __bf16* kbuf    = (__bf16*)carve(MR * DM * 2);
    __bf16* vbuf    = (__bf16*)carve(MR * DM * 2);
    __bf16* attnb   = (__bf16*)carve(MR * DM * 2);
    __bf16* hbuf    = (__bf16*)carve(MR * DM * 2);
    __bf16* c1buf   = (__bf16*)carve(MR * (DM / 2) * 2);
    __bf16* biasb   = (__bf16*)carve((size_t)NB * SS * SS * NH * 2);
    float*  scoresb = (float*) carve((size_t)NB * NH * SS * SS * 4);
    __bf16* probsb  = (__bf16*)carve((size_t)NB * NH * SS * SS * 2);

    const dim3 blk(256);
    init_kernel<<<dim3((unsigned)((MR * DM + 255) / 256)), blk, 0, stream>>>(msa, x_f32,
                                                                             x_bf, outp);

    for (int l = 0; l < NL; ++l) {
        const size_t lDD = (size_t)l * DM * DM;
        const size_t lD  = (size_t)l * DM;

        pair_bias_kernel<<<dim3((NB * SS * SS) / 128), blk, 0, stream>>>(pair, Wpb, l,
                                                                         biasb);

        gemm_bf16_kernel<0, 0, 0><<<dim3(6, 8), blk, 0, stream>>>(
            x_bf, Wq + lDD, bq + lD, nullptr, qbuf, (int)MR, DM, DM);
        gemm_bf16_kernel<0, 0, 0><<<dim3(6, 8), blk, 0, stream>>>(
            x_bf, Wk + lDD, bk + lD, nullptr, kbuf, (int)MR, DM, DM);
        gemm_bf16_kernel<0, 0, 0><<<dim3(6, 8), blk, 0, stream>>>(
            x_bf, Wv + lDD, bv + lD, nullptr, vbuf, (int)MR, DM, DM);

        scores_kernel<<<dim3(SS / 16, SS / 128, NB * NH), blk, 0, stream>>>(
            qbuf, kbuf, biasb, bpb, l, scoresb);

        softmax_kernel<<<dim3((NB * NH * SS) / 8), blk, 0, stream>>>(scoresb, probsb);

        attnv_kernel<<<dim3(SS / 128, 1, NB * NH), blk, 0, stream>>>(probsb, vbuf, attnb);

        // x = x + attn_out @ Wo + bo (residual in f32, mirror to bf16)
        gemm_bf16_kernel<1, 0, 1><<<dim3(6, 8), blk, 0, stream>>>(
            attnb, Wo + lDD, bo + lD, x_f32, x_bf, (int)MR, DM, DM);
        // h = relu(x @ Wb1 + bb1)
        gemm_bf16_kernel<0, 1, 0><<<dim3(6, 8), blk, 0, stream>>>(
            x_bf, Wb1, bb1, nullptr, hbuf, (int)MR, DM, DM);
        coord_update_kernel<<<dim3(3), blk, 0, stream>>>(hbuf, Wb2, bb2, outp);
    }

    // confidence head
    gemm_bf16_kernel<0, 1, 0><<<dim3(6, 4), blk, 0, stream>>>(
        x_bf, Wc1, bc1, nullptr, c1buf, (int)MR, DM / 2, DM);
    confidence_kernel<<<dim3(3), blk, 0, stream>>>(c1buf, Wc2, bc2,
                                                   outp + (size_t)NB * SS * 3);
}
